// SparseAutoencoder_26585847562302
// MI455X (gfx1250) — matrix operand order repaired and checked by reading, not compiled, not run on hardware
//
#include <hip/hip_runtime.h>

#define NROW 4096
#define KDIM 4096
#define LDIM 32768
#define DDIM 4096
#define TOPK 64

typedef __attribute__((ext_vector_type(8)))  __bf16 v8bf;
typedef __attribute__((ext_vector_type(16))) __bf16 v16bf;
typedef __attribute__((ext_vector_type(8)))  float  v8f;

#if __has_builtin(__builtin_amdgcn_global_load_async_to_lds_b128) && \
    __has_builtin(__builtin_amdgcn_s_wait_asynccnt)
#define HAVE_ASYNC 1
typedef __attribute__((ext_vector_type(4))) int v4i;
typedef __attribute__((address_space(1))) v4i* as1_v4i;
typedef __attribute__((address_space(3))) v4i* as3_v4i;
#else
#define HAVE_ASYNC 0
#endif

static __device__ __forceinline__ unsigned short f2bf(float f) {
  unsigned int u = __float_as_uint(f);
  u += 0x7FFFu + ((u >> 16) & 1u);          // round-to-nearest-even
  return (unsigned short)(u >> 16);
}

// monotonic sortable key for f32 (ascending)
static __device__ __forceinline__ unsigned int fkey(float f) {
  unsigned int u = __float_as_uint(f);
  return (u & 0x80000000u) ? ~u : (u | 0x80000000u);
}

// ---------------------------------------------------------------- conversions

__global__ __launch_bounds__(256) void prep_xc_kernel(
    const float* __restrict__ x, const float* __restrict__ b_pre,
    unsigned short* __restrict__ xc) {
  const size_t total = (size_t)NROW * KDIM;
  for (size_t i = (size_t)blockIdx.x * blockDim.x + threadIdx.x; i < total;
       i += (size_t)gridDim.x * blockDim.x)
    xc[i] = f2bf(x[i] - b_pre[i & (KDIM - 1)]);
}

// Tiled transpose + f32->bf16: in[R][Ccols] f32  ->  out[Ccols][R] bf16.
// Grid: (Ccols/32, R/32), block 256 (= 32x8).
__global__ __launch_bounds__(256) void tr_cvt_kernel(
    const float* __restrict__ in, unsigned short* __restrict__ out,
    int R, int Ccols) {
  __shared__ float t[32][33];
  const int tx = threadIdx.x & 31;
  const int ty = threadIdx.x >> 5;
  const int c0 = blockIdx.x * 32;
  const int r0 = blockIdx.y * 32;
#pragma unroll
  for (int j = 0; j < 4; ++j)
    t[ty + j * 8][tx] = in[(size_t)(r0 + ty + j * 8) * Ccols + c0 + tx];
  __syncthreads();
#pragma unroll
  for (int j = 0; j < 4; ++j)
    out[(size_t)(c0 + ty + j * 8) * R + r0 + tx] = f2bf(t[tx][ty + j * 8]);
}

// ------------------------------------------------------- bf16 WMMA GEMM
// C[m,n] = sum_k A[m,k]*Bt[k,n] (+ bias[n]); A:[M][K] bf16, Bt:[K][Nc] bf16.
// 128x128 block tile, K-step 32, double-buffered LDS filled with
// GLOBAL_LOAD_ASYNC_TO_LDS_B128 (ASYNCcnt) when available.
// 8 waves: 2 along M (64 rows) x 4 along N (32 cols); 4x2 wmma accums each.

#define BM 128
#define BN 128
#define BK 32

__global__ __launch_bounds__(256) void gemm_bf16_kernel(
    const unsigned short* __restrict__ A, const unsigned short* __restrict__ Bt,
    const float* __restrict__ bias, float* __restrict__ C, int Kc, int Nc) {
  __shared__ __align__(16) unsigned short sA[2][BM * BK];  // [row][k]
  __shared__ __align__(16) unsigned short sB[2][BK * BN];  // [k][col]

  const int tid  = threadIdx.x;
  const int lane = tid & 31;
  const int wave = tid >> 5;
  const int wm   = wave & 1;   // 0..1 -> 64 M-rows each
  const int wn   = wave >> 1;  // 0..3 -> 32 N-cols each
  const int m0   = blockIdx.y * BM;
  const int n0   = blockIdx.x * BN;

  const int mrow = lane & 15;        // M (or N) index within a 16-tile
  const int klo  = (lane >> 4) * 8;  // per-lane K sub-block select (A layout)

  v8f acc[4][2];
#pragma unroll
  for (int nt = 0; nt < 2; ++nt) {
    const float bv = bias ? bias[n0 + wn * 32 + nt * 16 + mrow] : 0.f;
#pragma unroll
    for (int mt = 0; mt < 4; ++mt)
#pragma unroll
      for (int j = 0; j < 8; ++j) acc[mt][nt][j] = bv;
  }

  // per-thread tile staging: 2x A-chunks + 2x B-chunks of 16B each
  auto issueTile = [&](int k0, int buf) {
#pragma unroll
    for (int h = 0; h < 2; ++h) {
      const int q    = tid + h * 256;
      const int arow = q >> 2, akc = (q & 3) * 8;
      const unsigned short* ga = A + (size_t)(m0 + arow) * Kc + k0 + akc;
      unsigned short*       la = &sA[buf][arow * BK + akc];
      const int brow = q >> 4, bnc = (q & 15) * 8;
      const unsigned short* gb = Bt + (size_t)(k0 + brow) * Nc + n0 + bnc;
      unsigned short*       lb = &sB[buf][brow * BN + bnc];
#if HAVE_ASYNC
      __builtin_amdgcn_global_load_async_to_lds_b128((as1_v4i)ga, (as3_v4i)la, 0, 0);
      __builtin_amdgcn_global_load_async_to_lds_b128((as1_v4i)gb, (as3_v4i)lb, 0, 0);
#else
      *(uint4*)la = *(const uint4*)ga;
      *(uint4*)lb = *(const uint4*)gb;
#endif
    }
  };

  const int ntile = Kc / BK;
  issueTile(0, 0);
  for (int t = 0; t < ntile; ++t) {
    const int cur = t & 1;
    if (t + 1 < ntile) issueTile((t + 1) * BK, cur ^ 1);
#if HAVE_ASYNC
    // this wave issues 4 async ops per tile; async loads complete in order,
    // so <=4 outstanding means tile t has fully landed in LDS
    if (t + 1 < ntile) __builtin_amdgcn_s_wait_asynccnt(4);
    else               __builtin_amdgcn_s_wait_asynccnt(0);
#endif
    __syncthreads();

    // A frag: lane<16 holds M=lane, K {0..7,16..23}; lane>=16: K {8..15,24..31}
    v16bf afrag[4], bfrag[2];
#pragma unroll
    for (int mt = 0; mt < 4; ++mt) {
      const __bf16* p =
          (const __bf16*)(&sA[cur][(wm * 64 + mt * 16 + mrow) * BK + klo]);
      v8bf lo = *(const v8bf*)p;
      v8bf hi = *(const v8bf*)(p + 16);
#pragma unroll
      for (int j = 0; j < 8; ++j) { afrag[mt][j] = lo[j]; afrag[mt][j + 8] = hi[j]; }
    }
    // B frag: lane holds col N=lane&15, K {klo..klo+7, klo+16..klo+23}
#pragma unroll
    for (int nt = 0; nt < 2; ++nt) {
      const __bf16* p = (const __bf16*)(&sB[cur][klo * BN + wn * 32 + nt * 16 + mrow]);
#pragma unroll
      for (int j = 0; j < 8; ++j) { bfrag[nt][j] = p[j * BN]; bfrag[nt][j + 8] = p[(j + 16) * BN]; }
    }
#pragma unroll
    for (int mt = 0; mt < 4; ++mt)
#pragma unroll
      for (int nt = 0; nt < 2; ++nt)
        acc[mt][nt] = __builtin_amdgcn_wmma_f32_16x16x32_bf16(
            false, afrag[mt], false, bfrag[nt], (short)0, acc[mt][nt], false, false);
    __syncthreads();
  }

  // C layout: VGPR v -> M = v + 8*(lane>=16); N = lane&15
  const int chalf = (lane >> 4) * 8;
#pragma unroll
  for (int mt = 0; mt < 4; ++mt)
#pragma unroll
    for (int nt = 0; nt < 2; ++nt) {
      const size_t base = (size_t)(m0 + wm * 64 + mt * 16 + chalf) * Nc +
                          (n0 + wn * 32 + nt * 16 + mrow);
#pragma unroll
      for (int v = 0; v < 8; ++v) C[base + (size_t)v * Nc] = acc[mt][nt][v];
    }
}

// ---------------------------------------------------- exact top-64 per row
// 3-level radix select on the sortable key (bits 31:20, 19:9, 8:0).

__global__ __launch_bounds__(256) void topk_kernel(
    const float* __restrict__ pre, float* __restrict__ z,
    int* __restrict__ tki, float* __restrict__ tkv) {
  const int m = blockIdx.x, tid = threadIdx.x;
  const float* row = pre + (size_t)m * LDIM;
  __shared__ unsigned int hist[4096];
  __shared__ int s_b1, s_r1, s_b2, s_r2, s_b3, s_r3;
  __shared__ unsigned int s_tie, s_cnt;

  for (int i = tid; i < 4096; i += 256) hist[i] = 0;
  __syncthreads();
  for (int i = tid; i < LDIM; i += 256) atomicAdd(&hist[fkey(row[i]) >> 20], 1u);
  __syncthreads();
  if (tid == 0) {
    int cum = 0, b = 4095;
    for (; b >= 0; --b) { int c = (int)hist[b]; if (cum + c >= TOPK) break; cum += c; }
    s_b1 = b; s_r1 = TOPK - cum;
  }
  __syncthreads();
  const unsigned int b1 = (unsigned)s_b1; const int r1 = s_r1;

  for (int i = tid; i < 2048; i += 256) hist[i] = 0;
  __syncthreads();
  for (int i = tid; i < LDIM; i += 256) {
    unsigned int k = fkey(row[i]);
    if ((k >> 20) == b1) atomicAdd(&hist[(k >> 9) & 2047u], 1u);
  }
  __syncthreads();
  if (tid == 0) {
    int cum = 0, b = 2047;
    for (; b >= 0; --b) { int c = (int)hist[b]; if (cum + c >= r1) break; cum += c; }
    s_b2 = b; s_r2 = r1 - cum;
  }
  __syncthreads();
  const unsigned int b2 = (unsigned)s_b2; const int r2 = s_r2;

  for (int i = tid; i < 512; i += 256) hist[i] = 0;
  __syncthreads();
  for (int i = tid; i < LDIM; i += 256) {
    unsigned int k = fkey(row[i]);
    if ((k >> 20) == b1 && ((k >> 9) & 2047u) == b2) atomicAdd(&hist[k & 511u], 1u);
  }
  __syncthreads();
  if (tid == 0) {
    int cum = 0, b = 511;
    for (; b >= 0; --b) { int c = (int)hist[b]; if (cum + c >= r2) break; cum += c; }
    s_b3 = b; s_r3 = r2 - cum; s_tie = 0; s_cnt = 0;
  }
  __syncthreads();
  const unsigned int T = (b1 << 20) | (b2 << 9) | (unsigned)s_b3;
  const int r3 = s_r3;
  if (tid < TOPK) { tki[m * TOPK + tid] = 0; tkv[m * TOPK + tid] = 0.f; }
  __syncthreads();

  for (int i = tid; i < LDIM; i += 256) {
    const float v = row[i];
    const unsigned int k = fkey(v);
    float o = 0.f;
    bool take = (k > T);
    if (!take && k == T) take = ((int)atomicAdd(&s_tie, 1u) < r3);
    if (take) {
      o = v > 0.f ? v : 0.f;  // ReLU of kept value
      const unsigned int p = atomicAdd(&s_cnt, 1u);
      if (p < TOPK) { tki[m * TOPK + p] = i; tkv[m * TOPK + p] = o; }
    }
    z[(size_t)m * LDIM + i] = o;
  }
}

// ------------------------------- sparse decode + residual target + ||res||

__global__ __launch_bounds__(256) void xhat_kernel(
    const float* __restrict__ x, const int* __restrict__ tki,
    const float* __restrict__ tkv, const float* __restrict__ W_dec,
    const float* __restrict__ b_dec, const float* __restrict__ b_pre,
    float* __restrict__ xhat, float* __restrict__ tgt, float* __restrict__ resn) {
  const int m = blockIdx.x, tid = threadIdx.x;
  __shared__ float sv[TOPK];
  __shared__ int   si[TOPK];
  __shared__ float red[256];
  if (tid < TOPK) { sv[tid] = tkv[m * TOPK + tid]; si[tid] = tki[m * TOPK + tid]; }
  __syncthreads();

  float acc[DDIM / 256];
#pragma unroll
  for (int t = 0; t < DDIM / 256; ++t) acc[t] = 0.f;

  for (int j = 0; j < TOPK; ++j) {
    const float v = sv[j];
    if (v == 0.f) continue;  // ReLU'd away
    const int id = si[j];
#pragma unroll
    for (int t = 0; t < DDIM / 256; ++t) {
      const int d = t * 256 + tid;
      acc[t] += v * W_dec[(size_t)d * LDIM + id];
    }
  }

  float ss = 0.f;
#pragma unroll
  for (int t = 0; t < DDIM / 256; ++t) {
    const int d = t * 256 + tid;
    const float xh = acc[t] + b_dec[d] + b_pre[d];
    const float xv = x[(size_t)m * DDIM + d];
    xhat[(size_t)m * DDIM + d] = xh;
    const float r = xv - xh;
    tgt[(size_t)m * DDIM + d] = 0.5f * r;
    ss += r * r;
  }
  red[tid] = ss; __syncthreads();
  for (int s = 128; s > 0; s >>= 1) { if (tid < s) red[tid] += red[tid + s]; __syncthreads(); }
  if (tid == 0) resn[m] = sqrtf(red[0]);
}

// --------------------------------------------------------- ghost path

__global__ void zero_nd_kernel(int* nd) { *nd = 0; }

__global__ __launch_bounds__(256) void dmask_kernel(
    const int* __restrict__ steps, const int* __restrict__ thr,
    int* __restrict__ dm, int* __restrict__ nd) {
  const int i = blockIdx.x * blockDim.x + threadIdx.x;
  if (i < LDIM) {
    const int d = (steps[i] >= thr[0]) ? 1 : 0;
    dm[i] = d;
    if (d) atomicAdd(nd, 1);
  }
}

__global__ __launch_bounds__(256) void gsum_kernel(
    const float* __restrict__ pre, const int* __restrict__ dm, float* __restrict__ gs) {
  const int m = blockIdx.x, tid = threadIdx.x;
  __shared__ float red[256];
  const float* row = pre + (size_t)m * LDIM;
  float s = 0.f;
  for (int l = tid; l < LDIM; l += 256)
    if (dm[l]) s += __expf(row[l]);
  red[tid] = s; __syncthreads();
  for (int st = 128; st > 0; st >>= 1) { if (tid < st) red[tid] += red[tid + st]; __syncthreads(); }
  if (tid == 0) gs[m] = fmaxf(red[0], 1e-8f);
}

__global__ __launch_bounds__(256) void gmat_kernel(
    const float* __restrict__ pre, const int* __restrict__ dm,
    const float* __restrict__ gs, unsigned short* __restrict__ G) {
  const size_t total = (size_t)NROW * LDIM;
  for (size_t i = (size_t)blockIdx.x * blockDim.x + threadIdx.x; i < total;
       i += (size_t)gridDim.x * blockDim.x) {
    const int l = (int)(i & (LDIM - 1));
    const size_t m = i >> 15;
    const float e = dm[l] ? __expf(pre[i]) : 0.f;
    G[i] = f2bf(e / gs[m]);
  }
}

__global__ __launch_bounds__(256) void gfin_kernel(
    float* __restrict__ xg, const float* __restrict__ resn,
    const int* __restrict__ nd, float* __restrict__ o_nd) {
  const int m = blockIdx.x, tid = threadIdx.x;
  __shared__ float red[256];
  __shared__ float s_scale;
  float ss = 0.f;
  for (int d = tid; d < DDIM; d += 256) {
    const float v = xg[(size_t)m * DDIM + d];
    ss += v * v;
  }
  red[tid] = ss; __syncthreads();
  for (int st = 128; st > 0; st >>= 1) { if (tid < st) red[tid] += red[tid + st]; __syncthreads(); }
  if (tid == 0) {
    const float gn = fmaxf(sqrtf(red[0]), 1e-8f);
    s_scale = resn[m] / gn * 0.5f;
  }
  __syncthreads();
  const float sc = s_scale;
  for (int d = tid; d < DDIM; d += 256) xg[(size_t)m * DDIM + d] *= sc;
  if (m == 0 && tid == 0) *o_nd = (float)(*nd);
}

// ---------------------------------------------------------------- launch

extern "C" void kernel_launch(void* const* d_in, const int* in_sizes, int n_in,
                              void* d_out, int out_size, void* d_ws, size_t ws_size,
                              hipStream_t stream) {
  (void)in_sizes; (void)n_in; (void)out_size; (void)ws_size;
  const float* x     = (const float*)d_in[0];
  const float* W_enc = (const float*)d_in[1];
  const float* b_enc = (const float*)d_in[2];
  const float* W_dec = (const float*)d_in[3];
  const float* b_dec = (const float*)d_in[4];
  const float* b_pre = (const float*)d_in[5];
  const int*   steps = (const int*)d_in[6];
  const int*   thr   = (const int*)d_in[7];

  float* out = (float*)d_out;
  const size_t NL = (size_t)NROW * LDIM;
  const size_t ND = (size_t)NROW * DDIM;
  float* o_pre  = out;            // pre_acts [N, L]
  float* o_z    = out + NL;       // z        [N, L]
  float* o_xhat = out + 2 * NL;   // x_hat    [N, D]
  float* o_xg   = o_xhat + ND;    // x_ghost_sc [N, D]
  float* o_tgt  = o_xg + ND;      // ghost_tgt  [N, D]
  float* o_nd   = o_tgt + ND;     // n_dead (scalar)

  char* p = (char*)d_ws;
  unsigned short* ws_xc = (unsigned short*)p; p += (size_t)NROW * KDIM * 2;
  unsigned short* ws_we = (unsigned short*)p; p += (size_t)LDIM * KDIM * 2;  // W_enc^T bf16 [K][L]
  unsigned short* ws_wd = (unsigned short*)p; p += (size_t)DDIM * LDIM * 2;  // W_dec^T bf16 [L][D]
  unsigned short* ws_g  = (unsigned short*)p; p += (size_t)NROW * LDIM * 2;
  int*   tki  = (int*)p;   p += (size_t)NROW * TOPK * 4;
  float* tkv  = (float*)p; p += (size_t)NROW * TOPK * 4;
  float* resn = (float*)p; p += (size_t)NROW * 4;
  float* gs   = (float*)p; p += (size_t)NROW * 4;
  int*   dm   = (int*)p;   p += (size_t)LDIM * 4;
  int*   nd   = (int*)p;   p += 16;

  // bf16 operand staging (weights transposed so GEMM tiles are contiguous)
  prep_xc_kernel<<<4096, 256, 0, stream>>>(x, b_pre, ws_xc);
  tr_cvt_kernel<<<dim3(KDIM / 32, LDIM / 32), 256, 0, stream>>>(
      W_enc, ws_we, LDIM, KDIM);   // [L][K] -> [K][L]
  tr_cvt_kernel<<<dim3(LDIM / 32, DDIM / 32), 256, 0, stream>>>(
      W_dec, ws_wd, DDIM, LDIM);   // [D][L] -> [L][D]

  // encoder GEMM: pre_acts = x_c @ W_enc.T + b_enc  (WMMA bf16, async LDS fill)
  gemm_bf16_kernel<<<dim3(LDIM / BN, NROW / BM), 256, 0, stream>>>(
      ws_xc, ws_we, b_enc, o_pre, KDIM, LDIM);

  // exact top-64 + dense z + compact lists
  topk_kernel<<<NROW, 256, 0, stream>>>(o_pre, o_z, tki, tkv);

  // sparse decode + ghost_tgt + residual norms
  xhat_kernel<<<NROW, 256, 0, stream>>>(x, tki, tkv, W_dec, b_dec, b_pre,
                                        o_xhat, o_tgt, resn);

  // ghost path
  zero_nd_kernel<<<1, 1, 0, stream>>>(nd);
  dmask_kernel<<<LDIM / 256, 256, 0, stream>>>(steps, thr, dm, nd);
  gsum_kernel<<<NROW, 256, 0, stream>>>(o_pre, dm, gs);
  gmat_kernel<<<8192, 256, 0, stream>>>(o_pre, dm, gs, ws_g);

  // ghost GEMM: x_ghost_raw = G @ W_dec.T  (WMMA bf16, K = 32768)
  gemm_bf16_kernel<<<dim3(DDIM / BN, NROW / BM), 256, 0, stream>>>(
      ws_g, ws_wd, nullptr, o_xg, LDIM, DDIM);

  gfin_kernel<<<NROW, 256, 0, stream>>>(o_xg, resn, nd, o_nd);
}
